// FrameDenoiser2_87935160418335
// MI455X (gfx1250) — compile-verified, matrix operand order checked by reading, not run on hardware
//
#include <hip/hip_runtime.h>

// ---------------- problem sizes ----------------
#define NN   4096
#define LL   512
#define NB   8
#define CS   256
#define CVc  16
#define CZc  128
#define NH   8
#define KNNK 50
#define ESP  (NN*KNNK)
#define ESE  (NN*4)
#define HT   64
#define SHID 128

typedef __attribute__((ext_vector_type(16))) __bf16 v16bf;
typedef __attribute__((ext_vector_type(8)))  float  v8f;

struct __align__(16) U4 { unsigned a, b, c, d; };
struct __align__(8)  U2 { unsigned a, b; };
struct __align__(16) F4 { float x, y, z, w; };
union Frag { v16bf v; U4 q[2]; };

__device__ __forceinline__ unsigned short f2bf(float f) {
  union { float f; unsigned u; } c; c.f = f;
  unsigned r = c.u + 0x7fffu + ((c.u >> 16) & 1u);
  return (unsigned short)(r >> 16);
}

__device__ __forceinline__ v16bf ldfrag(const unsigned short* rowp, int half) {
  Frag f;
  f.q[0] = *reinterpret_cast<const U4*>(rowp + 8 * half);
  f.q[1] = *reinterpret_cast<const U4*>(rowp + 16 + 8 * half);
  return f.v;
}

// ---------------- WMMA GEMM with gathered 3-way concat A ----------------
// C[M x Nc] = act( concat(X0[I0[r]](w0), X1[I1[r]](w1), X2[I2[r]](w2)) @ W[Kc x Nc] + bias )
// I* are always valid index arrays (identity handled via iota). Row pitch of X_i is w_i.
// All segment boundaries (w0, w0+w1) must be multiples of 4 (true for every call here).
#define BM 64
#define BN 64
#define BK 64

__global__ __launch_bounds__(256)
void k_gemm(const float* X0, const int* I0, int w0,
            const float* X1, const int* I1, int w1,
            const float* X2, const int* I2, int w2,
            const float* W, const float* bias,
            float* C, int M, int Kc, int Nc, int act)
{
  __shared__ __align__(16) unsigned short As[BM][BK];
  __shared__ __align__(16) unsigned short Bs[BN][BK];
  int tid = threadIdx.x;
  int m0 = blockIdx.x * BM, n0 = blockIdx.y * BN;
  int wv = tid >> 5, lane = tid & 31;
  int half = lane >> 4, l16 = lane & 15;
  int wm = (wv & 3) * 16;    // 4 row tiles of 16
  int wn = (wv >> 2) * 32;   // 2 col groups of 32
  int w01 = w0 + w1;

  // hoisted per-row gather state: each thread owns 4 A-chunks (4 rows)
  int am[4], akc[4];
  const float* a0[4]; const float* a1[4]; const float* a2[4];
  bool rok[4];
#pragma unroll
  for (int it = 0; it < 4; ++it) {
    int i = tid + it * 256;            // 1024 chunks (64 rows x 16 chunks)
    am[it] = i >> 4;
    akc[it] = (i & 15) * 4;
    int row = m0 + am[it];
    rok[it] = row < M;
    int rr = rok[it] ? row : 0;
    a0[it] = X0 + (long)I0[rr] * w0;
    a1[it] = X1 + (long)I1[rr] * w1;
    a2[it] = X2 + (long)I2[rr] * w2;
  }

  v8f acc0 = {}, acc1 = {};
  for (int k0 = 0; k0 < Kc; k0 += BK) {
    // ---- A tile 64x64: gathered concat, 4-wide chunks, no loop-carried index loads ----
#pragma unroll
    for (int it = 0; it < 4; ++it) {
      int kc = akc[it];
      int gk = k0 + kc;
      const float* p = a0[it] + gk;
      if (gk >= w0)  p = a1[it] + (gk - w0);
      if (gk >= w01) p = a2[it] + (gk - w01);
      float v0 = 0.f, v1 = 0.f, v2 = 0.f, v3 = 0.f;
      if (rok[it] && gk + 3 < Kc) {
        F4 f = *reinterpret_cast<const F4*>(p);
        v0 = f.x; v1 = f.y; v2 = f.z; v3 = f.w;
      } else if (rok[it] && gk < Kc) {
        v0 = p[0];
        if (gk + 1 < Kc) v1 = p[1];
        if (gk + 2 < Kc) v2 = p[2];
        if (gk + 3 < Kc) v3 = p[3];
      }
      U2 pk;
      pk.a = (unsigned)f2bf(v0) | ((unsigned)f2bf(v1) << 16);
      pk.b = (unsigned)f2bf(v2) | ((unsigned)f2bf(v3) << 16);
      *reinterpret_cast<U2*>(&As[am[it]][kc]) = pk;
    }
    // ---- B tile: W[k0..+63][n0..+63] -> Bs[n][k], 4-wide k-chunks ----
#pragma unroll
    for (int it = 0; it < 4; ++it) {
      int i = tid + it * 256;            // 1024 chunks (64 n x 16 k-chunks)
      int n = i >> 4, kc = (i & 15) * 4;
      int gn = n0 + n, gk = k0 + kc;
      float v0 = 0.f, v1 = 0.f, v2 = 0.f, v3 = 0.f;
      if (gn < Nc) {
        const float* p = W + (long)gk * Nc + gn;
        if (gk + 3 < Kc) {
          v0 = p[0]; v1 = p[Nc]; v2 = p[2 * Nc]; v3 = p[3 * Nc];
        } else if (gk < Kc) {
          v0 = p[0];
          if (gk + 1 < Kc) v1 = p[Nc];
          if (gk + 2 < Kc) v2 = p[2 * Nc];
          if (gk + 3 < Kc) v3 = p[3 * Nc];
        }
      }
      U2 pk;
      pk.a = (unsigned)f2bf(v0) | ((unsigned)f2bf(v1) << 16);
      pk.b = (unsigned)f2bf(v2) | ((unsigned)f2bf(v3) << 16);
      *reinterpret_cast<U2*>(&Bs[n][kc]) = pk;
    }
    __syncthreads();
#pragma unroll
    for (int sub = 0; sub < 2; ++sub) {
      v16bf fa  = ldfrag(&As[wm + l16][sub * 32], half);
      v16bf fb0 = ldfrag(&Bs[wn + l16][sub * 32], half);
      v16bf fb1 = ldfrag(&Bs[wn + 16 + l16][sub * 32], half);
      acc0 = __builtin_amdgcn_wmma_f32_16x16x32_bf16(false, fa, false, fb0, (short)0, acc0, false, false);
      acc1 = __builtin_amdgcn_wmma_f32_16x16x32_bf16(false, fa, false, fb1, (short)0, acc1, false, false);
    }
    __syncthreads();
  }
#pragma unroll
  for (int a = 0; a < 2; ++a) {
    int col = n0 + wn + a * 16 + l16;
    if (col < Nc) {
      float bv = bias ? bias[col] : 0.f;
      v8f acc = a ? acc1 : acc0;
#pragma unroll
      for (int r = 0; r < 8; ++r) {
        int row = m0 + wm + r + 8 * half;
        if (row < M) {
          float v = acc[r] + bv;
          if (act == 1) v = fmaxf(v, 0.f);
          else if (act == 2) v = 1.f / (1.f + __expf(-v));
          C[(long)row * Nc + col] = v;
        }
      }
    }
  }
}

// ---------------- small kernels ----------------
__global__ void k_iota(int* p, int total) {
  int i = blockIdx.x * blockDim.x + threadIdx.x;
  if (i < total) p[i] = i;
}
__global__ void k_copyf(const float* a, float* b, long total) {
  long i = (long)blockIdx.x * blockDim.x + threadIdx.x;
  if (i < total) b[i] = a[i];
}
__global__ void k_xvalid(const unsigned char* xm, float* xv) {
  int n = blockIdx.x * blockDim.x + threadIdx.x;
  if (n < NN) xv[n] = xm[n] ? 0.f : 1.f;
}
__global__ void k_mulrows(float* x, const float* sc, int C, long total) {
  long i = (long)blockIdx.x * blockDim.x + threadIdx.x;
  if (i < total) x[i] *= sc[i / C];
}
__global__ void k_mulrows_to(const float* x, const float* sc, float* o, int C, long total) {
  long i = (long)blockIdx.x * blockDim.x + threadIdx.x;
  if (i < total) o[i] = x[i] * sc[i / C];
}

__global__ __launch_bounds__(256) void k_center(const float* trans, float* t, float* center) {
  __shared__ float red[256][3];
  int b = blockIdx.x, tid = threadIdx.x;
  float s0 = 0, s1 = 0, s2 = 0;
  for (int r = tid; r < LL; r += 256) {
    const float* p = trans + (long)(b * LL + r) * 3;
    s0 += p[0]; s1 += p[1]; s2 += p[2];
  }
  red[tid][0] = s0; red[tid][1] = s1; red[tid][2] = s2;
  __syncthreads();
  for (int off = 128; off; off >>= 1) {
    if (tid < off) { red[tid][0] += red[tid+off][0]; red[tid][1] += red[tid+off][1]; red[tid][2] += red[tid+off][2]; }
    __syncthreads();
  }
  float c0 = red[0][0] / LL, c1 = red[0][1] / LL, c2 = red[0][2] / LL;
  if (tid == 0) { center[b*3+0] = c0; center[b*3+1] = c1; center[b*3+2] = c2; }
  for (int r = tid; r < LL; r += 256) {
    int n = b * LL + r;
    t[n*3+0] = (trans[n*3+0] - c0) * 0.1f;
    t[n*3+1] = (trans[n*3+1] - c1) * 0.1f;
    t[n*3+2] = (trans[n*3+2] - c2) * 0.1f;
  }
}

__global__ __launch_bounds__(256) void k_knn(const float* t, const unsigned char* xm, int* dstE, int* srcE) {
  __shared__ float d2[LL];
  __shared__ float rv[256];
  __shared__ int   ri[256];
  int n = blockIdx.x, tid = threadIdx.x;
  int base = (n / LL) * LL;
  float tx = t[n*3], ty = t[n*3+1], tz = t[n*3+2];
  for (int j = tid; j < LL; j += 256) {
    int g = base + j;
    float dx = t[g*3] - tx, dy = t[g*3+1] - ty, dz = t[g*3+2] - tz;
    float d = dx*dx + dy*dy + dz*dz;
    if (g == n || xm[g]) d = 3.0e38f;
    d2[j] = d;
  }
  __syncthreads();
  for (int sel = 0; sel < KNNK; ++sel) {
    float bv = d2[tid]; int bi = tid;
    float v2 = d2[tid + 256];
    if (v2 < bv) { bv = v2; bi = tid + 256; }
    rv[tid] = bv; ri[tid] = bi;
    __syncthreads();
    for (int off = 128; off; off >>= 1) {
      if (tid < off && rv[tid+off] < rv[tid]) { rv[tid] = rv[tid+off]; ri[tid] = ri[tid+off]; }
      __syncthreads();
    }
    if (tid == 0) {
      int j = ri[0];
      srcE[n * KNNK + sel] = base + j;
      dstE[n * KNNK + sel] = n;
      d2[j] = 3.0e38f;
    }
    __syncthreads();
  }
}

__global__ void k_seqidx(int* dstS, int* srcS) {
  int i = blockIdx.x * blockDim.x + threadIdx.x;
  if (i >= ESE) return;
  const int offs[4] = {-2, -1, 1, 2};
  int n = i >> 2;
  int j = n + offs[i & 3];
  bool ok = (j >= 0) && (j < NN) && ((j / LL) == (n / LL));
  dstS[i] = n;
  srcS[i] = ok ? j : n;
}

__global__ void k_edge_feats(const float* t, const int* dst, const int* src, float* z, int E) {
  long i = (long)blockIdx.x * blockDim.x + threadIdx.x;
  if (i >= (long)E * CZc) return;
  int e = (int)(i / CZc), c = (int)(i % CZc);
  int d = dst[e], s = src[e];
  float out;
  if (c < 64) {
    float dx = t[d*3]-t[s*3], dy = t[d*3+1]-t[s*3+1], dz = t[d*3+2]-t[s*3+2];
    float dist = sqrtf(dx*dx + dy*dy + dz*dz);
    float mu = 20.f * c / 63.f;
    float u = (dist - mu) / (20.f / 64.f);
    out = __expf(-u * u);
  } else {
    float rel = (float)(d - s);
    int j = (c - 64) & 31;
    float freq = __expf(-((float)(2*j) / 64.f) * logf(10000.f));
    float ang = rel * freq;
    out = (c < 96) ? __cosf(ang) : __sinf(ang);
  }
  z[(long)e * CZc + c] = out;
}

__global__ void k_rbf_time(const float* ts, float* ft) {
  long i = (long)blockIdx.x * blockDim.x + threadIdx.x;
  if (i >= (long)NN * HT) return;
  int n = (int)(i / HT), c = (int)(i % HT);
  float mu = (float)c / 63.f;
  float u = (ts[n] - mu) / (1.f / 64.f);
  ft[i] = __expf(-u * u);
}

__global__ void k_posemb(float* pe) {
  long i = (long)blockIdx.x * blockDim.x + threadIdx.x;
  if (i >= (long)NN * CS) return;
  int n = (int)(i / CS), c = (int)(i % CS);
  int j = c & 127;
  float freq = __expf(-((float)(2*j) / 256.f) * logf(10000.f));
  float ang = (float)n * freq;
  pe[i] = (c < 128) ? __cosf(ang) : __sinf(ang);
}

__global__ void k_toglobal(float* p, const float* R, const float* t) {
  int i = blockIdx.x * blockDim.x + threadIdx.x;
  if (i >= NN * 64) return;
  int n = i >> 6;
  float* q = p + (long)n * 192 + (i & 63) * 3;
  float x = q[0], y = q[1], z = q[2];
  const float* r = R + n * 9;
  q[0] = r[0]*x + r[1]*y + r[2]*z + t[n*3+0];
  q[1] = r[3]*x + r[4]*y + r[5]*z + t[n*3+1];
  q[2] = r[6]*x + r[7]*y + r[8]*z + t[n*3+2];
}

__global__ void k_logits(const float* q, const float* k, const float* qpg, const float* kpg,
                         const float* zb, const float* gamma, const int* dst, const int* src,
                         float* logit, int E) {
  long i = (long)blockIdx.x * blockDim.x + threadIdx.x;
  if (i >= (long)E * NH) return;
  int e = (int)(i / NH), h = (int)(i % NH);
  int dn = dst[e], sn = src[e];
  const float* qd = q + (long)dn * 128 + h * 16;
  const float* ks = k + (long)sn * 128 + h * 16;
  float qk = 0;
  for (int c = 0; c < 16; ++c) qk += qd[c] * ks[c];
  qk *= 0.25f;
  const float* qp = qpg + (long)dn * 192 + h * 24;
  const float* kp = kpg + (long)sn * 192 + h * 24;
  float pd = 0;
  for (int c = 0; c < 24; ++c) { float d = qp[c] - kp[c]; pd += d * d; }
  float g = gamma[h];
  float sp = (g > 20.f) ? g : log1pf(__expf(g));
  logit[i] = (qk + zb[(long)e * NH + h] - 0.5f * sp * pd) * 0.57735026919f;
}

__global__ __launch_bounds__(256) void k_softmax(float* logit, int Kk) {
  int n = blockIdx.x;
  int h = threadIdx.x >> 5, lane = threadIdx.x & 31;
  long e0 = (long)n * Kk;
  float m = -3.0e38f;
  for (int kk = lane; kk < Kk; kk += 32) m = fmaxf(m, logit[(e0 + kk) * NH + h]);
  for (int off = 16; off; off >>= 1) m = fmaxf(m, __shfl_xor(m, off, 32));
  float sum = 0.f;
  for (int kk = lane; kk < Kk; kk += 32) {
    float ex = __expf(logit[(e0 + kk) * NH + h] - m);
    logit[(e0 + kk) * NH + h] = ex;
    sum += ex;
  }
  for (int off = 16; off; off >>= 1) sum += __shfl_xor(sum, off, 32);
  float inv = 1.f / (sum + 1e-9f);
  for (int kk = lane; kk < Kk; kk += 32) logit[(e0 + kk) * NH + h] *= inv;
}

__global__ __launch_bounds__(256) void k_agg(const float* a, const float* vs, const float* zv,
                                             const float* vpg, const int* src, const float* R,
                                             const float* t, float* feats, int Kk) {
  __shared__ float aw[KNNK * NH];
  __shared__ float op[192];
  int n = blockIdx.x, tid = threadIdx.x;
  long e0 = (long)n * Kk;
  for (int i = tid; i < Kk * NH; i += 256) aw[i] = a[e0 * NH + i];
  __syncthreads();
  float* fr = feats + (long)n * 512;
  if (tid < 128) {
    int h = tid >> 4, c = tid & 15;
    float acc = 0;
    for (int kk = 0; kk < Kk; ++kk) {
      int s = src[e0 + kk];
      acc += aw[kk * NH + h] * vs[(long)s * 128 + h * 16 + c];
    }
    fr[tid] = acc;
  } else {
    int c2 = tid - 128;
    int h = c2 >> 4, c = c2 & 15;
    float acc = 0;
    for (int kk = 0; kk < Kk; ++kk) acc += aw[kk * NH + h] * zv[(e0 + kk) * 16 + c];
    fr[128 + c2] = acc;
  }
  if (tid < 192) {
    int h = tid / 24;
    float acc = 0;
    for (int kk = 0; kk < Kk; ++kk) {
      int s = src[e0 + kk];
      acc += aw[kk * NH + h] * vpg[(long)s * 192 + tid];
    }
    op[tid] = acc;
  }
  __syncthreads();
  if (tid < 64) {
    float dx = op[tid*3+0] - t[n*3+0];
    float dy = op[tid*3+1] - t[n*3+1];
    float dz = op[tid*3+2] - t[n*3+2];
    const float* r = R + n * 9;
    float ox = r[0]*dx + r[3]*dy + r[6]*dz;
    float oy = r[1]*dx + r[4]*dy + r[7]*dz;
    float oz = r[2]*dx + r[5]*dy + r[8]*dz;
    fr[256 + tid*3 + 0] = ox;
    fr[256 + tid*3 + 1] = oy;
    fr[256 + tid*3 + 2] = oz;
    fr[448 + tid] = sqrtf(ox*ox + oy*oy + oz*oz + 1e-8f);
  }
}

__global__ void k_vupd(const float* feats, const float* Wvo, const float* v, const float* Wvmix, float* dv) {
  int i = blockIdx.x * blockDim.x + threadIdx.x;
  if (i >= NN * 48) return;
  int n = i / 48, rem = i % 48, cc = rem / 3, ii = rem % 3;
  const float* opl = feats + (long)n * 512 + 256;
  float acc = 0;
  for (int hp = 0; hp < 64; ++hp) acc += opl[hp * 3 + ii] * Wvo[hp * 16 + cc];
  if (Wvmix)
    for (int c = 0; c < 16; ++c) acc += v[(long)n * 48 + c * 3 + ii] * Wvmix[c * 16 + cc];
  dv[i] = acc;
}

__global__ __launch_bounds__(256) void k_sln(float* s, const float* ds, const float* xv) {
  __shared__ float r1[256], r2[256];
  int n = blockIdx.x, tid = threadIdx.x;
  float x = s[(long)n * CS + tid] + ds[(long)n * CS + tid] * (xv ? xv[n] : 1.f);
  r1[tid] = x; r2[tid] = x * x;
  __syncthreads();
  for (int off = 128; off; off >>= 1) {
    if (tid < off) { r1[tid] += r1[tid+off]; r2[tid] += r2[tid+off]; }
    __syncthreads();
  }
  float m = r1[0] / CS;
  float var = r2[0] / CS - m * m;
  s[(long)n * CS + tid] = (x - m) * rsqrtf(var + 1e-5f);
}

__global__ __launch_bounds__(128) void k_ln128(float* x) {
  __shared__ float r1[128], r2[128];
  long n = blockIdx.x; int tid = threadIdx.x;
  float v = x[n * CZc + tid];
  r1[tid] = v; r2[tid] = v * v;
  __syncthreads();
  for (int off = 64; off; off >>= 1) {
    if (tid < off) { r1[tid] += r1[tid+off]; r2[tid] += r2[tid+off]; }
    __syncthreads();
  }
  float m = r1[0] / CZc;
  float var = r2[0] / CZc - m * m;
  x[n * CZc + tid] = (v - m) * rsqrtf(var + 1e-5f);
}

__global__ __launch_bounds__(64) void k_vln(float* v, const float* dv, const float* xv, int add) {
  __shared__ float nv[48];
  __shared__ float rms;
  int n = blockIdx.x, tid = threadIdx.x;
  if (tid < 48) {
    float x = (add ? v[(long)n * 48 + tid] : 0.f) + dv[(long)n * 48 + tid] * xv[n];
    nv[tid] = x;
  }
  __syncthreads();
  if (tid == 0) {
    float s = 0;
    for (int i = 0; i < 48; ++i) s += nv[i] * nv[i];
    rms = sqrtf(s / 16.f + 1e-5f);
  }
  __syncthreads();
  if (tid < 48) v[(long)n * 48 + tid] = nv[tid] / rms;
}

__global__ void k_vnorm(const float* v, float* vn) {
  int i = blockIdx.x * blockDim.x + threadIdx.x;
  if (i >= NN * CVc) return;
  const float* p = v + (long)i * 3;
  vn[i] = sqrtf(p[0]*p[0] + p[1]*p[1] + p[2]*p[2] + 1e-8f);
}

__global__ void k_local_vupd(const float* v, const float* W, const float* gate, float* dv) {
  int i = blockIdx.x * blockDim.x + threadIdx.x;
  if (i >= NN * 48) return;
  int n = i / 48, rem = i % 48, d = rem / 3, ii = rem % 3;
  float acc = 0;
  for (int c = 0; c < 16; ++c) acc += v[(long)n * 48 + c * 3 + ii] * W[c * 16 + d];
  dv[i] = gate[n * 16 + d] * acc;
}

__global__ void k_vmix(const float* v, const float* W, float* dv) {
  int i = blockIdx.x * blockDim.x + threadIdx.x;
  if (i >= NN * 48) return;
  int n = i / 48, rem = i % 48, d = rem / 3, ii = rem % 3;
  float acc = 0;
  for (int c = 0; c < 16; ++c) acc += v[(long)n * 48 + c * 3 + ii] * W[c * 16 + d];
  dv[i] = acc;
}

__global__ void k_bbcompose(const float* bb6, const float* v, const float* nm, const float* Wv,
                            float* R, float* t) {
  int n = blockIdx.x * blockDim.x + threadIdx.x;
  if (n >= NN) return;
  float nmv = nm[n];
  float upd[6];
  for (int kk = 0; kk < 2; ++kk)
    for (int ii = 0; ii < 3; ++ii) {
      float acc = 0;
      for (int c = 0; c < 16; ++c) acc += v[(long)n * 48 + c * 3 + ii] * nmv * Wv[c * 2 + kk];
      upd[kk * 3 + ii] = acc;
    }
  for (int j = 0; j < 6; ++j) upd[j] = (bb6[n * 6 + j] + upd[j]) * nmv;
  float b = upd[0], c = upd[1], d = upd[2];
  float qn = rsqrtf(1.f + b*b + c*c + d*d);
  float w = qn, x = b * qn, y = c * qn, zq = d * qn;
  float Ru[9] = {1-2*(y*y+zq*zq), 2*(x*y-w*zq),   2*(x*zq+w*y),
                 2*(x*y+w*zq),    1-2*(x*x+zq*zq),2*(y*zq-w*x),
                 2*(x*zq-w*y),    2*(y*zq+w*x),   1-2*(x*x+y*y)};
  float rl[9], Ro[9];
  for (int i = 0; i < 9; ++i) rl[i] = R[n * 9 + i];
  for (int i = 0; i < 3; ++i)
    for (int j = 0; j < 3; ++j) {
      float acc = 0;
      for (int kk = 0; kk < 3; ++kk) acc += rl[i*3+kk] * Ru[kk*3+j];
      Ro[i*3+j] = acc;
    }
  for (int i = 0; i < 3; ++i) {
    float acc = 0;
    for (int j = 0; j < 3; ++j) acc += rl[i*3+j] * upd[3+j];
    t[n*3+i] += acc;
  }
  for (int i = 0; i < 9; ++i) R[n * 9 + i] = Ro[i];
}

__device__ __constant__ float BBI[12] = {-0.525f, 1.363f, 0.f,  0.f, 0.f, 0.f,
                                          1.526f, 0.f,    0.f,  2.153f, -1.062f, 0.f};

__global__ void k_final(const float* t, const float* center, const float* R,
                        float* tout, float* bbout) {
  int n = blockIdx.x * blockDim.x + threadIdx.x;
  if (n >= NN) return;
  int b = n / LL;
  float to[3];
  for (int i = 0; i < 3; ++i) {
    to[i] = t[n*3+i] * 10.f + center[b*3+i];
    tout[n*3+i] = to[i];
  }
  const float* r = R + n * 9;
  for (int a = 0; a < 4; ++a)
    for (int i = 0; i < 3; ++i)
      bbout[(n*4+a)*3+i] = r[i*3+0]*BBI[a*3+0] + r[i*3+1]*BBI[a*3+1] + r[i*3+2]*BBI[a*3+2] + to[i];
}

// ---------------- host driver ----------------
extern "C" void kernel_launch(void* const* d_in, const int* in_sizes, int n_in,
                              void* d_out, int out_size, void* d_ws, size_t ws_size,
                              hipStream_t stream) {
  (void)in_sizes; (void)n_in; (void)out_size; (void)ws_size;
  int pi = 0;
  auto nf = [&]() { return (const float*)d_in[pi++]; };

  struct Attn { const float *Wb,*Wk,*Wkp,*Wo,*Wq,*Wqp,*Wv,*Wvmix,*Wvo,*Wvp,*Wzv,*bo,*gamma; };
  struct Layer {
    Attn aseq, asp;
    const float *bbW,*bbWv,*bbb;
    const float *etW1,*etW2,*etb1,*etb2;
    const float *loW1,*loWg,*loWs,*loWv,*lob1;
    const float *ntW1,*ntW2,*ntWv,*ntb1,*ntb2;
    const float *seW1,*seW2,*seb1,*seb2;
  };
  // JAX pytree flatten: sorted dict keys. top: embed, layers, time.
  const float* embW = nf(); const float* embb = nf();
  Layer Lp[4];
  auto fillA = [&](Attn& a, bool mix) {
    a.Wb = nf(); a.Wk = nf(); a.Wkp = nf(); a.Wo = nf(); a.Wq = nf(); a.Wqp = nf(); a.Wv = nf();
    a.Wvmix = mix ? nf() : nullptr;
    a.Wvo = nf(); a.Wvp = nf(); a.Wzv = nf(); a.bo = nf(); a.gamma = nf();
  };
  for (int l = 0; l < 4; ++l) {
    fillA(Lp[l].aseq, l != 0);
    fillA(Lp[l].asp, true);
    Lp[l].bbW = nf(); Lp[l].bbWv = nf(); Lp[l].bbb = nf();
    Lp[l].etW1 = nf(); Lp[l].etW2 = nf(); Lp[l].etb1 = nf(); Lp[l].etb2 = nf();
    Lp[l].loW1 = nf(); Lp[l].loWg = nf(); Lp[l].loWs = nf(); Lp[l].loWv = nf(); Lp[l].lob1 = nf();
    Lp[l].ntW1 = nf(); Lp[l].ntW2 = nf(); Lp[l].ntWv = nf(); Lp[l].ntb1 = nf(); Lp[l].ntb2 = nf();
    Lp[l].seW1 = nf(); Lp[l].seW2 = nf(); Lp[l].seb1 = nf(); Lp[l].seb2 = nf();
  }
  const float* tW1 = nf(); const float* tW2 = nf(); const float* tb1 = nf(); const float* tb2 = nf();
  const float* trans = nf();
  const float* rots  = nf();
  const float* ts    = nf();
  const unsigned char* xm = (const unsigned char*)d_in[pi++];
  const float* nm    = nf();

  // workspace carve
  float* w = (float*)d_ws;
  size_t off = 0;
  auto alloc = [&](size_t nfl) { float* p = w + off; off += nfl; return p; };
  float* t_buf  = alloc((size_t)NN * 3);
  float* center = alloc((size_t)NB * 3 + 2);
  float* Rbuf   = alloc((size_t)NN * 9);
  float* z      = alloc((size_t)ESP * CZc);
  float* zs     = alloc((size_t)ESE * CZc);
  float* sbuf   = alloc((size_t)NN * CS);
  float* vbuf   = alloc((size_t)NN * 48);
  float* scr    = alloc((size_t)ESP * CZc);
  float* qb     = alloc((size_t)NN * 128);
  float* kb     = alloc((size_t)NN * 128);
  float* vsb    = alloc((size_t)NN * 128);
  float* qpb    = alloc((size_t)NN * 192);
  float* kpb    = alloc((size_t)NN * 192);
  float* vpb    = alloc((size_t)NN * 192);
  float* feats  = alloc((size_t)NN * 512);
  float* hloc   = alloc((size_t)NN * SHID);
  float* gate   = alloc((size_t)NN * CVc);
  float* ds_buf = alloc((size_t)NN * CS);
  float* dv_buf = alloc((size_t)NN * 48);
  float* ftb    = alloc((size_t)NN * HT);
  float* etb    = alloc((size_t)NN * HT);
  float* pe     = alloc((size_t)NN * CS);
  float* smask  = alloc((size_t)NN * CS);
  float* bb6    = alloc((size_t)NN * 6 + 2);
  float* vn     = alloc((size_t)NN * CVc);
  float* xvalid = alloc((size_t)NN);
  int* dstE = (int*)(w + off); off += ESP;
  int* srcE = (int*)(w + off); off += ESP;
  int* dstS = (int*)(w + off); off += ESE;
  int* srcS = (int*)(w + off); off += ESE;
  int* iota = (int*)(w + off); off += ESP;

  auto gemm = [&](const float* X0, const int* I0, int w0,
                  const float* X1, const int* I1, int w1,
                  const float* X2, const int* I2, int w2,
                  const float* W, const float* bias, float* C,
                  int M, int Kc, int Nc, int act) {
    dim3 g((M + BM - 1) / BM, (Nc + BN - 1) / BN);
    k_gemm<<<g, dim3(256), 0, stream>>>(X0, I0 ? I0 : iota, w0,
                                        X1 ? X1 : X0, I1 ? I1 : iota, w1,
                                        X2 ? X2 : X0, I2 ? I2 : iota, w2,
                                        W, bias, C, M, Kc, Nc, act);
  };
  auto gemm1 = [&](const float* A, const float* W, const float* b, float* C, int M, int Kc, int Nc, int act) {
    gemm(A, nullptr, Kc, nullptr, nullptr, 0, nullptr, nullptr, 0, W, b, C, M, Kc, Nc, act);
  };
  auto GB = [](long total) { return dim3((unsigned)((total + 255) / 256)); };

  // ---- preamble ----
  k_iota<<<GB(ESP), 256, 0, stream>>>(iota, ESP);
  k_xvalid<<<GB(NN), 256, 0, stream>>>(xm, xvalid);
  k_copyf<<<GB((long)NN * 9), 256, 0, stream>>>(rots, Rbuf, (long)NN * 9);
  k_center<<<NB, 256, 0, stream>>>(trans, t_buf, center);
  k_knn<<<NN, 256, 0, stream>>>(t_buf, xm, dstE, srcE);
  k_seqidx<<<GB(ESE), 256, 0, stream>>>(dstS, srcS);
  k_edge_feats<<<GB((long)ESP * CZc), 256, 0, stream>>>(t_buf, dstE, srcE, z, ESP);
  k_edge_feats<<<GB((long)ESE * CZc), 256, 0, stream>>>(t_buf, dstS, srcS, zs, ESE);
  k_rbf_time<<<GB((long)NN * HT), 256, 0, stream>>>(ts, ftb);
  gemm1(ftb, tW1, tb1, hloc, NN, HT, SHID, 1);
  gemm1(hloc, tW2, tb2, etb, NN, SHID, HT, 1);
  k_posemb<<<GB((long)NN * CS), 256, 0, stream>>>(pe);
  gemm(pe, nullptr, CS, etb, nullptr, HT, nm, nullptr, 1, embW, embb, sbuf, NN, CS + HT + 1, CS, 0);

  auto run_attn = [&](const Attn& a, const float* zE, const int* dE, const int* sE,
                      int E, int Kk, bool hasv) {
    gemm1(sbuf, a.Wq,  nullptr, qb,  NN, CS, 128, 0);
    gemm1(sbuf, a.Wk,  nullptr, kb,  NN, CS, 128, 0);
    gemm1(sbuf, a.Wv,  nullptr, vsb, NN, CS, 128, 0);
    gemm1(sbuf, a.Wqp, nullptr, qpb, NN, CS, 192, 0);
    gemm1(sbuf, a.Wkp, nullptr, kpb, NN, CS, 192, 0);
    gemm1(sbuf, a.Wvp, nullptr, vpb, NN, CS, 192, 0);
    k_toglobal<<<GB(NN * 64), 256, 0, stream>>>(qpb, Rbuf, t_buf);
    k_toglobal<<<GB(NN * 64), 256, 0, stream>>>(kpb, Rbuf, t_buf);
    k_toglobal<<<GB(NN * 64), 256, 0, stream>>>(vpb, Rbuf, t_buf);
    float* zbv = scr;
    float* zvv = scr + (size_t)E * 8;
    float* logit = scr + (size_t)E * 24;
    gemm1(zE, a.Wb,  nullptr, zbv, E, CZc, 8, 0);
    gemm1(zE, a.Wzv, nullptr, zvv, E, CZc, 16, 0);
    k_logits<<<GB((long)E * NH), 256, 0, stream>>>(qb, kb, qpb, kpb, zbv, a.gamma, dE, sE, logit, E);
    k_softmax<<<NN, 256, 0, stream>>>(logit, Kk);
    k_agg<<<NN, 256, 0, stream>>>(logit, vsb, zvv, vpb, sE, Rbuf, t_buf, feats, Kk);
    gemm1(feats, a.Wo, a.bo, ds_buf, NN, 512, CS, 0);
    k_vupd<<<GB(NN * 48), 256, 0, stream>>>(feats, a.Wvo, vbuf, hasv ? a.Wvmix : nullptr, dv_buf);
  };

  // ---- layers ----
  for (int l = 0; l < 4; ++l) {
    const Layer& P = Lp[l];
    // spatial edge transition
    gemm(sbuf, dstE, CS, sbuf, srcE, CS, z, nullptr, CZc, P.etW1, P.etb1, scr, ESP, 2*CS + CZc, CZc, 1);
    gemm1(scr, P.etW2, P.etb2, z, ESP, CZc, CZc, 0);
    k_ln128<<<ESP, 128, 0, stream>>>(z);
    // sequence edge transition
    gemm(sbuf, dstS, CS, sbuf, srcS, CS, zs, nullptr, CZc, P.seW1, P.seb1, scr, ESE, 2*CS + CZc, CZc, 1);
    gemm1(scr, P.seW2, P.seb2, zs, ESE, CZc, CZc, 0);
    k_ln128<<<ESE, 128, 0, stream>>>(zs);
    // sequence attention
    run_attn(P.aseq, zs, dstS, srcS, ESE, 4, l > 0);
    k_sln<<<NN, 256, 0, stream>>>(sbuf, ds_buf, xvalid);
    k_vln<<<NN, 64, 0, stream>>>(vbuf, dv_buf, xvalid, l > 0 ? 1 : 0);
    // spatial attention
    run_attn(P.asp, z, dstE, srcE, ESP, KNNK, true);
    k_sln<<<NN, 256, 0, stream>>>(sbuf, ds_buf, xvalid);
    k_vln<<<NN, 64, 0, stream>>>(vbuf, dv_buf, xvalid, 1);
    // local update
    k_vnorm<<<GB(NN * CVc), 256, 0, stream>>>(vbuf, vn);
    gemm(sbuf, nullptr, CS, vn, nullptr, CVc, nullptr, nullptr, 0, P.loW1, P.lob1, hloc, NN, CS + CVc, SHID, 1);
    gemm1(hloc, P.loWs, nullptr, ds_buf, NN, SHID, CS, 0);
    gemm1(hloc, P.loWg, nullptr, gate, NN, SHID, CVc, 2);
    k_local_vupd<<<GB(NN * 48), 256, 0, stream>>>(vbuf, P.loWv, gate, dv_buf);
    k_sln<<<NN, 256, 0, stream>>>(sbuf, ds_buf, xvalid);
    k_vln<<<NN, 64, 0, stream>>>(vbuf, dv_buf, xvalid, 1);
    // node transition
    gemm1(sbuf, P.ntW1, P.ntb1, feats, NN, CS, CS, 1);
    gemm1(feats, P.ntW2, P.ntb2, ds_buf, NN, CS, CS, 0);
    k_sln<<<NN, 256, 0, stream>>>(sbuf, ds_buf, nullptr);
    k_vmix<<<GB(NN * 48), 256, 0, stream>>>(vbuf, P.ntWv, dv_buf);
    k_mulrows<<<GB((long)NN * CS), 256, 0, stream>>>(sbuf, xvalid, CS, (long)NN * CS);
    k_mulrows_to<<<GB((long)NN * 48), 256, 0, stream>>>(dv_buf, xvalid, vbuf, 48, (long)NN * 48);
    // backbone update + compose
    k_mulrows_to<<<GB((long)NN * CS), 256, 0, stream>>>(sbuf, nm, smask, CS, (long)NN * CS);
    gemm1(smask, P.bbW, P.bbb, bb6, NN, CS, 6, 0);
    k_bbcompose<<<GB(NN), 256, 0, stream>>>(bb6, vbuf, nm, P.bbWv, Rbuf, t_buf);
  }

  // ---- outputs: s, t_out, bb ----
  float* out = (float*)d_out;
  k_copyf<<<GB((long)NN * CS), 256, 0, stream>>>(sbuf, out, (long)NN * CS);
  k_final<<<GB(NN), 256, 0, stream>>>(t_buf, center, Rbuf, out + (size_t)NN * CS,
                                      out + (size_t)NN * CS + (size_t)NN * 3);
}
